// AnalyticalCatastropheDetector_43224550867220
// MI455X (gfx1250) — matrix operand order repaired and checked by reading, not compiled, not run on hardware
//
#include <hip/hip_runtime.h>
#include <hip/hip_bf16.h>

typedef __attribute__((ext_vector_type(16))) _Float16 v16h;
typedef __attribute__((ext_vector_type(8)))  _Float16 v8h;
typedef __attribute__((ext_vector_type(8)))  float    v8f;

__device__ __forceinline__ float sigm_f(float x) { return 1.0f / (1.0f + __expf(-x)); }

// Branch-free exact-style gelu: erf via Abramowitz-Stegun 7.1.26 (|err|<=1.5e-7).
// Avoids ocml erff's exec-mask branches between WMMAs.
__device__ __forceinline__ float gelu_f(float x) {
  float ax = fabsf(x) * 0.70710678118654752f;           // |x|/sqrt(2)
  float t  = __builtin_amdgcn_rcpf(fmaf(0.3275911f, ax, 1.0f));
  float p  = fmaf(1.061405429f, t, -1.453152027f);
  p = fmaf(p, t, 1.421413741f);
  p = fmaf(p, t, -0.284496736f);
  p = fmaf(p, t, 0.254829592f);
  p = p * t;
  float e  = __expf(-ax * ax);
  float erf_abs = fmaf(-p, e, 1.0f);                    // erf(|x|/sqrt2)
  float erf_v   = copysignf(erf_abs, x);
  return 0.5f * x * (1.0f + erf_v);
}

// ws layout (floats): [0,2048) Wfused 64x32 row-major (cols 29..31 zero),
//                     [2048,2080) fused bias (29 valid, pad 0),
//                     [2080,2087) softmax(g2_weights)
__global__ void acd_fuse_setup_kernel(const float* __restrict__ W2,
                                      const float* __restrict__ b2,
                                      const float* __restrict__ Wh,
                                      const float* __restrict__ bh,
                                      const float* __restrict__ g2w,
                                      float* __restrict__ ws) {
  int t = threadIdx.x;
  for (int idx = t; idx < 64 * 32; idx += blockDim.x) {
    int r = idx >> 5, n = idx & 31;
    float s = 0.0f;
    if (n < 29) {
      for (int k = 0; k < 32; ++k) s = fmaf(W2[r * 32 + k], Wh[k * 29 + n], s);
    }
    ws[idx] = s;
  }
  if (t < 32) {
    float s = 0.0f;
    if (t < 29) {
      s = bh[t];
      for (int k = 0; k < 32; ++k) s = fmaf(b2[k], Wh[k * 29 + t], s);
    }
    ws[2048 + t] = s;
  }
  if (t == 0) {
    float mx = g2w[0];
    for (int i = 1; i < 7; ++i) mx = fmaxf(mx, g2w[i]);
    float e[7], sum = 0.0f;
    for (int i = 0; i < 7; ++i) { e[i] = __expf(g2w[i] - mx); sum += e[i]; }
    float inv = 1.0f / sum;
    for (int i = 0; i < 7; ++i) ws[2080 + i] = e[i] * inv;
  }
}

__launch_bounds__(256, 1)
__global__ void acd_main_kernel(const float* __restrict__ emb,
                                const float* __restrict__ W1,
                                const float* __restrict__ b1,
                                const float* __restrict__ ws,
                                float* __restrict__ out,
                                int nRows, int numBT) {
  const int lane = threadIdx.x & 31;
  const int wave = threadIdx.x >> 5;
  const int m    = lane & 15;   // matrix row (A) / col (B,C)
  const int half = lane >> 4;   // 0: lanes 0-15, 1: lanes 16-31

  // per-wave private LDS scratch (no cross-wave sharing -> no barriers)
  __shared__ __align__(16) _Float16 sh_h[8][16][72];  // gelu output, padded pitch
  __shared__ __align__(16) float    sh_c[8][16][36];  // final C, 16B-aligned rows

  const float* Wf  = ws;
  const float* bfv = ws + 2048;
  const float* wsm = ws + 2080;

  // ---- Build register-resident B fragments (ISA 16-bit B layout:
  //      element i -> K = 32*kt + 16*(i/8) + 8*half + i%8, N = 16*nt + m) ----
  v16h w1f[2][4];
#pragma unroll
  for (int kt = 0; kt < 2; ++kt)
#pragma unroll
    for (int nt = 0; nt < 4; ++nt)
#pragma unroll
      for (int i = 0; i < 16; ++i)
        w1f[kt][nt][i] = (_Float16)W1[(kt * 32 + (i & 8) * 2 + half * 8 + (i & 7)) * 64 + nt * 16 + m];

  v16h wff[2][2];
#pragma unroll
  for (int kt = 0; kt < 2; ++kt)
#pragma unroll
    for (int nt = 0; nt < 2; ++nt)
#pragma unroll
      for (int i = 0; i < 16; ++i)
        wff[kt][nt][i] = (_Float16)Wf[(kt * 32 + (i & 8) * 2 + half * 8 + (i & 7)) * 32 + nt * 16 + m];

  float b1c[4];
#pragma unroll
  for (int nt = 0; nt < 4; ++nt) b1c[nt] = b1[nt * 16 + m];
  float bfc[2];
#pragma unroll
  for (int nt = 0; nt < 2; ++nt) bfc[nt] = bfv[nt * 16 + m];
  float w7[7];
#pragma unroll
  for (int i = 0; i < 7; ++i) w7[i] = wsm[i];

  // ---- Persistent grid-stride loop over block-tiles (128 rows each) ----
  for (int bt = blockIdx.x; bt < numBT; bt += gridDim.x) {
    const int row0 = (bt * 8 + wave) * 16;
    if (row0 >= nRows) continue;

    // prefetch next iteration's embedding rows into cache
    {
      int nbt = bt + gridDim.x;
      if (nbt < numBT) {
        int nrow0 = (nbt * 8 + wave) * 16;
        if (nrow0 < nRows)
          __builtin_prefetch(emb + (size_t)(nrow0 + m) * 64 + half * 32, 0, 0);
      }
    }

    // ---- Load A fragments (fp32 -> f16), ISA 16-bit A layout:
    //      elem i<8  -> K = 32*kt + 8*half + i
    //      elem i>=8 -> K = 32*kt + 16 + 8*half + (i-8) ----
    const int rowm = (row0 + m < nRows) ? (row0 + m) : (nRows - 1);
    const float* arow = emb + (size_t)rowm * 64;
    v16h afrag[2];
#pragma unroll
    for (int kt = 0; kt < 2; ++kt) {
      const float4 p0 = *(const float4*)(arow + kt * 32 + half * 8);
      const float4 p1 = *(const float4*)(arow + kt * 32 + half * 8 + 4);
      const float4 p2 = *(const float4*)(arow + kt * 32 + 16 + half * 8);
      const float4 p3 = *(const float4*)(arow + kt * 32 + 16 + half * 8 + 4);
      afrag[kt][0]  = (_Float16)p0.x; afrag[kt][1]  = (_Float16)p0.y;
      afrag[kt][2]  = (_Float16)p0.z; afrag[kt][3]  = (_Float16)p0.w;
      afrag[kt][4]  = (_Float16)p1.x; afrag[kt][5]  = (_Float16)p1.y;
      afrag[kt][6]  = (_Float16)p1.z; afrag[kt][7]  = (_Float16)p1.w;
      afrag[kt][8]  = (_Float16)p2.x; afrag[kt][9]  = (_Float16)p2.y;
      afrag[kt][10] = (_Float16)p2.z; afrag[kt][11] = (_Float16)p2.w;
      afrag[kt][12] = (_Float16)p3.x; afrag[kt][13] = (_Float16)p3.y;
      afrag[kt][14] = (_Float16)p3.z; afrag[kt][15] = (_Float16)p3.w;
    }

    // ---- GEMM1 (16x64 @ 64x64) + branch-free gelu, write f16 tile to LDS ----
#pragma unroll
    for (int nt = 0; nt < 4; ++nt) {
      v8f acc;
#pragma unroll
      for (int r = 0; r < 8; ++r) acc[r] = b1c[nt];  // bias folded into C init
      acc = __builtin_amdgcn_wmma_f32_16x16x32_f16(false, afrag[0], false, w1f[0][nt],
                                                   (short)0, acc, false, false);
      acc = __builtin_amdgcn_wmma_f32_16x16x32_f16(false, afrag[1], false, w1f[1][nt],
                                                   (short)0, acc, false, false);
#pragma unroll
      for (int r = 0; r < 8; ++r)
        sh_h[wave][r + half * 8][nt * 16 + m] = (_Float16)gelu_f(acc[r]);
    }

    // ---- Reload gelu tile as A fragments (16B-aligned ds_load_b128) ----
    const _Float16* hrow = &sh_h[wave][m][0];
    v16h hfrag[2];
#pragma unroll
    for (int kt = 0; kt < 2; ++kt) {
      const v8h q0 = *(const v8h*)(hrow + kt * 32 + half * 8);
      const v8h q1 = *(const v8h*)(hrow + kt * 32 + 16 + half * 8);
#pragma unroll
      for (int i = 0; i < 8; ++i) { hfrag[kt][i] = q0[i]; hfrag[kt][i + 8] = q1[i]; }
    }

    // ---- GEMM2 (16x64 @ 64x32 fused W2@Wh), stash C rows in LDS ----
#pragma unroll
    for (int nt = 0; nt < 2; ++nt) {
      v8f acc;
#pragma unroll
      for (int r = 0; r < 8; ++r) acc[r] = bfc[nt];  // fused bias
      acc = __builtin_amdgcn_wmma_f32_16x16x32_f16(false, hfrag[0], false, wff[0][nt],
                                                   (short)0, acc, false, false);
      acc = __builtin_amdgcn_wmma_f32_16x16x32_f16(false, hfrag[1], false, wff[1][nt],
                                                   (short)0, acc, false, false);
#pragma unroll
      for (int r = 0; r < 8; ++r)
        sh_c[wave][r + half * 8][nt * 16 + m] = acc[r];
    }

    // ---- Per-row catastrophe risk heads: lanes 0-15 own one row each ----
    if (lane < 16 && row0 + lane < nRows) {
      const float* crow = &sh_c[wave][lane][0];
      float c[29];
#pragma unroll
      for (int q = 0; q < 7; ++q) {                 // 7x ds_load_b128
        const float4 v = *(const float4*)(crow + q * 4);
        c[q * 4 + 0] = v.x; c[q * 4 + 1] = v.y;
        c[q * 4 + 2] = v.z; c[q * 4 + 3] = v.w;
      }
      c[28] = crow[28];

      float rk[7];
      // fold (o=0)
      rk[0] = __expf(-fabsf(c[0])) * sigm_f(1.0f - fabsf(c[1]));
      // cusp (o=2)
      rk[1] = __expf(-fabsf(c[2] * c[2] - 3.0f * fabsf(c[3]))) * sigm_f(0.5f - fabsf(c[4]));
      // swallowtail (o=5)
      rk[2] = __expf(-(fabsf(c[7]) + fabsf(c[5] * c[5] - 4.0f * c[6]))) *
              sigm_f(0.3f - fabsf(c[8]));
      // butterfly (o=9)
      {
        float a = c[9], b = c[10], cc = c[11], d = c[12], e = c[13];
        float t1 = fabsf(d);
        float t2 = fabsf(cc - a * b * 0.5f);
        float t3 = fabsf(a * a * a - 4.5f * a * cc + 3.375f * b * b);
        rk[3] = __expf(-(t1 + t2 + t3) * (1.0f / 3.0f)) * sigm_f(0.2f - fabsf(e));
      }
      // hyperbolic umbilic (o=14)
      rk[4] = __expf(-fabsf(c[14] * c[14] * c[14] - 27.0f * c[15] * c[16]) * 0.1f) *
              sigm_f(fabsf(c[14]) - 0.5f);
      // elliptic umbilic (o=19)
      rk[5] = __expf(-fabsf(c[19] * c[19] * c[19] + 27.0f * c[20] * c[21]) * 0.1f) *
              sigm_f(-fabsf(c[19]) - 0.5f);
      // parabolic umbilic (o=24)
      rk[6] = __expf(-fabsf(c[24] * c[25] - c[26] * c[26] * 0.25f)) *
              sigm_f(fabsf(c[27]) + fabsf(c[28]) - 1.0f);

      float tot = 0.0f;
#pragma unroll
      for (int i = 0; i < 7; ++i) {
        rk[i] = fminf(fmaxf(rk[i], 0.0f), 1.0f);
        tot = fmaf(rk[i], w7[i], tot);
      }
      tot = fminf(fmaxf(tot, 0.0f), 1.0f);

      const int row = row0 + lane;
      out[row] = tot;
      float* rv = out + (size_t)nRows + (size_t)row * 7;
#pragma unroll
      for (int i = 0; i < 7; ++i) rv[i] = rk[i];
    }
  }
}

extern "C" void kernel_launch(void* const* d_in, const int* in_sizes, int n_in,
                              void* d_out, int out_size, void* d_ws, size_t ws_size,
                              hipStream_t stream) {
  const float* emb = (const float*)d_in[0];
  const float* W1  = (const float*)d_in[1];
  const float* b1  = (const float*)d_in[2];
  const float* W2  = (const float*)d_in[3];
  const float* b2  = (const float*)d_in[4];
  const float* Wh  = (const float*)d_in[5];
  const float* bh  = (const float*)d_in[6];
  const float* g2w = (const float*)d_in[7];
  float* out = (float*)d_out;
  float* ws  = (float*)d_ws;

  const int nRows = in_sizes[0] / 64;          // 1,000,000
  const int tiles = (nRows + 15) / 16;         // 62,500 (exact)
  const int numBT = (tiles + 7) / 8;           // block-tiles of 128 rows

  acd_fuse_setup_kernel<<<1, 256, 0, stream>>>(W2, b2, Wh, bh, g2w, ws);

  int grid = numBT < 1024 ? numBT : 1024;      // persistent blocks, grid-stride
  acd_main_kernel<<<grid, 256, 0, stream>>>(emb, W1, b1, ws, out, nRows, numBT);
}
